// LinearBin_49074296324136
// MI455X (gfx1250) — matrix operand order repaired and checked by reading, not compiled, not run on hardware
//
#include <hip/hip_runtime.h>

typedef __attribute__((ext_vector_type(16))) _Float16 v16h;
typedef __attribute__((ext_vector_type(8)))  _Float16 h8;
typedef __attribute__((ext_vector_type(8)))  float    v8f;
typedef __attribute__((ext_vector_type(4)))  float    f4;

#define M_TOT 8192
#define N_TOT 4096
#define K_TOT 4096
#define TM 128
#define TN 128
#define TK 32
#define KP 48            // padded row length in f16 (96B stride, 16B aligned)
#define THRESH 0.05f

union FragAB { v16h v; h8 h[2]; };

__device__ __forceinline__ _Float16 ternarize(float w) {
    float r = (w > THRESH) ? 1.0f : ((w < -THRESH) ? -1.0f : 0.0f);
    return (_Float16)r;
}

__global__ __launch_bounds__(256) void linbin_wmma_kernel(
    const float* __restrict__ X,      // [8192, 4096] f32
    const float* __restrict__ W,      // [4096, 4096] f32
    const float* __restrict__ bias,   // [4096] f32
    float* __restrict__ out)          // [8192, 4096] f32
{
    __shared__ _Float16 as[TM][KP];   // x tile (f16)
    __shared__ _Float16 bs[TN][KP];   // ternarized weight tile (f16)

    const int tid  = threadIdx.x;
    const int lane = tid & 31;
    const int wave = tid >> 5;
    const int lr   = lane & 15;
    const int half = lane >> 4;

    const int bm = blockIdx.y * TM;
    const int bn = blockIdx.x * TN;

    // 8 waves arranged 2 (M) x 4 (N); each wave computes 64x32
    const int wm = (wave & 1) * 64;
    const int wn = (wave >> 1) * 32;

    // global->LDS mapping: thread covers 16 consecutive k of one row
    const int lrow = tid >> 1;          // 0..127
    const int lcol = (tid & 1) * 16;    // 0 or 16

    const float* xp = X + (size_t)(bm + lrow) * K_TOT + lcol;
    const float* wp = W + (size_t)(bn + lrow) * K_TOT + lcol;

    const v8f vzero = {0.f, 0.f, 0.f, 0.f, 0.f, 0.f, 0.f, 0.f};
    v8f acc[4][2];
#pragma unroll
    for (int i = 0; i < 4; ++i)
#pragma unroll
        for (int j = 0; j < 2; ++j)
            acc[i][j] = vzero;

    f4 ra[4], rw[4];
    // prologue: stage k-block 0 into registers
#pragma unroll
    for (int u = 0; u < 4; ++u) {
        ra[u] = *(const f4*)(xp + u * 4);
        rw[u] = *(const f4*)(wp + u * 4);
    }

    const int NKB = K_TOT / TK;   // 128
    for (int kb = 0; kb < NKB; ++kb) {
        // ---- convert staged registers and store to LDS ----
        {
            h8 ha, hb, wa, wb;
#pragma unroll
            for (int e = 0; e < 4; ++e) {
                ha[e]     = (_Float16)ra[0][e];
                ha[e + 4] = (_Float16)ra[1][e];
                hb[e]     = (_Float16)ra[2][e];
                hb[e + 4] = (_Float16)ra[3][e];
                wa[e]     = ternarize(rw[0][e]);
                wa[e + 4] = ternarize(rw[1][e]);
                wb[e]     = ternarize(rw[2][e]);
                wb[e + 4] = ternarize(rw[3][e]);
            }
            *(h8*)&as[lrow][lcol]     = ha;
            *(h8*)&as[lrow][lcol + 8] = hb;
            *(h8*)&bs[lrow][lcol]     = wa;
            *(h8*)&bs[lrow][lcol + 8] = wb;
        }
        __syncthreads();

        // ---- prefetch next k-block into registers ----
        if (kb + 1 < NKB) {
            const float* xn = xp + (size_t)(kb + 1) * TK;
            const float* wn_ = wp + (size_t)(kb + 1) * TK;
#pragma unroll
            for (int u = 0; u < 4; ++u) {
                ra[u] = *(const f4*)(xn + u * 4);
                rw[u] = *(const f4*)(wn_ + u * 4);
            }
        }

        // ---- WMMA compute on current LDS tile ----
        // A fragment (16-bit 16x32): lanes 0-15: K{0..7,16..23}; lanes 16-31: K{8..15,24..31}
        const int kA = half * 8;
        // B fragment (32x16): lanes 0-15: K{0..7,16..23}; lanes 16-31: K{8..15,24..31}
        const int kB = half * 8;

        FragAB af[4], bf[2];
#pragma unroll
        for (int i = 0; i < 4; ++i) {
            const int row = wm + i * 16 + lr;
            af[i].h[0] = *(const h8*)&as[row][kA];
            af[i].h[1] = *(const h8*)&as[row][kA + 16];
        }
#pragma unroll
        for (int j = 0; j < 2; ++j) {
            const int col = wn + j * 16 + lr;
            bf[j].h[0] = *(const h8*)&bs[col][kB];
            bf[j].h[1] = *(const h8*)&bs[col][kB + 16];
        }

#pragma unroll
        for (int i = 0; i < 4; ++i)
#pragma unroll
            for (int j = 0; j < 2; ++j)
                acc[i][j] = __builtin_amdgcn_wmma_f32_16x16x32_f16(
                    /*neg_a=*/false, af[i].v,
                    /*neg_b=*/false, bf[j].v,
                    /*c_mod=*/(short)0, acc[i][j],
                    /*reuse_a=*/false, /*reuse_b=*/false);

        __syncthreads();
    }

    // ---- epilogue: add bias, store f32 ----
    // C/D layout: VGPR e, lanes 0-15 -> M=e, N=lane; lanes 16-31 -> M=8+e, N=lane-16
#pragma unroll
    for (int j = 0; j < 2; ++j) {
        const int col = bn + wn + j * 16 + lr;
        const float bv = bias[col];
#pragma unroll
        for (int i = 0; i < 4; ++i) {
            const int rbase = bm + wm + i * 16 + half * 8;
#pragma unroll
            for (int e = 0; e < 8; ++e) {
                out[(size_t)(rbase + e) * N_TOT + col] = acc[i][j][e] + bv;
            }
        }
    }
}

extern "C" void kernel_launch(void* const* d_in, const int* in_sizes, int n_in,
                              void* d_out, int out_size, void* d_ws, size_t ws_size,
                              hipStream_t stream) {
    (void)in_sizes; (void)n_in; (void)d_ws; (void)ws_size; (void)out_size;
    const float* X    = (const float*)d_in[0];
    const float* W    = (const float*)d_in[1];
    const float* bias = (const float*)d_in[2];
    float* out        = (float*)d_out;

    dim3 grid(N_TOT / TN, M_TOT / TM);   // (32, 64)
    linbin_wmma_kernel<<<grid, dim3(256), 0, stream>>>(X, W, bias, out);
}
